// SparseMLADecode_47682726920486
// MI455X (gfx1250) — matrix operand order repaired and checked by reading, not compiled, not run on hardware
//
#include <hip/hip_runtime.h>

#define BATCH   32
#define HEADS   128
#define SKV_N   8192
#define DIM_V   512
#define DTOT    576
#define TOPK_N  2048
#define QSTART  8191
#define NEG_BIG (-1e30f)

typedef __attribute__((ext_vector_type(16))) __bf16        v16bf;
typedef __attribute__((ext_vector_type(2)))  __bf16        v2bf;
typedef __attribute__((ext_vector_type(8)))  float         v8f;
typedef __attribute__((ext_vector_type(4)))  unsigned int  v4u;
typedef __attribute__((ext_vector_type(4)))  float         v4f;

__device__ __forceinline__ unsigned short f2bf(float f) {
    unsigned u = __float_as_uint(f);
    unsigned r = u + 0x7FFFu + ((u >> 16) & 1u);   // round-to-nearest-even
    return (unsigned short)(r >> 16);
}

// pack two f32 -> two bf16 in one dword (v_cvt_pk_bf16_f32 when available)
__device__ __forceinline__ unsigned pk_bf16(float a, float b) {
#if __has_builtin(__builtin_amdgcn_cvt_pk_bf16_f32)
    v2bf p = __builtin_amdgcn_cvt_pk_bf16_f32(a, b);
    return __builtin_bit_cast(unsigned, p);
#else
    return (unsigned)f2bf(a) | ((unsigned)f2bf(b) << 16);
#endif
}

// 16-lane xor-butterfly step via v_permlane16_b32 (VALU, no DS round-trip)
__device__ __forceinline__ float xor16(float v, int step) {
#if __has_builtin(__builtin_amdgcn_permlane16)
    int x = __builtin_bit_cast(int, v);
    int r;
    switch (step) {   // selector nibbles: sel[i] = i ^ mask (row-relative)
    case 0: r = __builtin_amdgcn_permlane16(x, x, 0x67452301, 0xEFCDAB89, false, false); break; // xor 1
    case 1: r = __builtin_amdgcn_permlane16(x, x, 0x54761032, 0xDCFE98BA, false, false); break; // xor 2
    case 2: r = __builtin_amdgcn_permlane16(x, x, 0x32107654, 0xBA98FEDC, false, false); break; // xor 4
    default:r = __builtin_amdgcn_permlane16(x, x, 0xFEDCBA98, 0x76543210, false, false); break; // xor 8
    }
    return __builtin_bit_cast(float, r);
#else
    return __shfl_xor(v, 1 << step, 32);
#endif
}

__device__ __forceinline__ float redmax16(float v) {
    v = fmaxf(v, xor16(v, 0));
    v = fmaxf(v, xor16(v, 1));
    v = fmaxf(v, xor16(v, 2));
    v = fmaxf(v, xor16(v, 3));
    return v;
}
__device__ __forceinline__ float redsum16(float v) {
    v += xor16(v, 0);
    v += xor16(v, 1);
    v += xor16(v, 2);
    v += xor16(v, 3);
    return v;
}

__global__ __launch_bounds__(128)
void sparse_mla_decode_kernel(const float* __restrict__ Q,
                              const float* __restrict__ KV,
                              const int*   __restrict__ Idx,
                              float*       __restrict__ Out) {
    const int htile  = blockIdx.x;        // 0..7  (16 heads each)
    const int b      = blockIdx.y;        // 0..31
    const int tid    = threadIdx.x;       // 0..127
    const int lane   = tid & 31;
    const int wave   = tid >> 5;          // 0..3, owns dims [wave*128, wave*128+128)
    const int ln16   = lane & 15;
    const bool hiH   = lane >= 16;

    __shared__ __align__(16) unsigned short q_lds[16 * DTOT];    // [head][dim] bf16
    __shared__ __align__(16) unsigned short k_lds[16 * DTOT];    // [key][dim]  bf16
    __shared__ __align__(16) unsigned short vt_lds[DIM_V * 16];  // [dim][key]  bf16
    __shared__ __align__(16) unsigned short p_lds[4 * 16 * 16];  // per-wave P  bf16

    const float scale = 1.0f / 24.0f;     // 1/sqrt(576)

    // ---- stage Q tile (pre-scaled) into LDS as bf16 ----
    {
        const int row = tid >> 3;                 // 16 rows, 8 threads per row
        const int c0  = (tid & 7) * 72;           // 72 floats per thread
        const float* qsrc = Q + (size_t)(b * HEADS + htile * 16 + row) * DTOT;
        #pragma unroll
        for (int i = 0; i < 72; i += 4) {
            v4f q4 = *(const v4f*)(qsrc + c0 + i);
            *(unsigned*)&q_lds[row * DTOT + c0 + i]     = pk_bf16(q4[0] * scale, q4[1] * scale);
            *(unsigned*)&q_lds[row * DTOT + c0 + i + 2] = pk_bf16(q4[2] * scale, q4[3] * scale);
        }
    }

    float m_r[8], l_r[8];
    v8f   o_acc[8];
    #pragma unroll
    for (int r = 0; r < 8; ++r) { m_r[r] = NEG_BIG; l_r[r] = 0.0f; o_acc[r] = (v8f)0.0f; }

    __syncthreads();

    for (int kb = 0; kb < TOPK_N / 16; ++kb) {
        // ---- gather 16 KV rows -> k_lds ([key][dim]) and vt_lds ([dim][key]) ----
        {
            const int row = tid >> 3;
            const int c0  = (tid & 7) * 72;
            const int key = Idx[b * TOPK_N + kb * 16 + row];
            const float* src = KV + ((size_t)b * SKV_N + (size_t)key) * DTOT + c0;

            // prefetch next block's rows toward L2/WGP$ while we convert this one
            if (kb + 1 < TOPK_N / 16) {
                const int nkey = Idx[b * TOPK_N + (kb + 1) * 16 + row];
                const char* nsrc = (const char*)(KV + ((size_t)b * SKV_N + (size_t)nkey) * DTOT + c0);
                __builtin_prefetch(nsrc, 0, 3);
                __builtin_prefetch(nsrc + 128, 0, 3);
                __builtin_prefetch(nsrc + 256, 0, 3);
            }

            #pragma unroll
            for (int i = 0; i < 72; i += 4) {
                const int d = c0 + i;
                v4f kv4 = *(const v4f*)(src + i);
                unsigned lo = pk_bf16(kv4[0], kv4[1]);
                unsigned hi = pk_bf16(kv4[2], kv4[3]);
                *(unsigned*)&k_lds[row * DTOT + d]     = lo;
                *(unsigned*)&k_lds[row * DTOT + d + 2] = hi;
                if (d < DIM_V) {
                    vt_lds[(d + 0) * 16 + row] = (unsigned short)(lo & 0xFFFFu);
                    vt_lds[(d + 1) * 16 + row] = (unsigned short)(lo >> 16);
                    vt_lds[(d + 2) * 16 + row] = (unsigned short)(hi & 0xFFFFu);
                    vt_lds[(d + 3) * 16 + row] = (unsigned short)(hi >> 16);
                }
            }
        }
        __syncthreads();

        // ---- scores S[16h x 16k] = Q (16x576) * Ksel^T : 18 bf16 WMMAs ----
        v8f s = (v8f)0.0f;
        #pragma unroll
        for (int c = 0; c < 18; ++c) {
            v16bf a, bm;
            // A = Q tile, 16x32 bf16 layout: lane<16 K={c*32+0..7, +16..23}, lane>=16 K={+8..15, +24..31}
            const unsigned short* qp = &q_lds[ln16 * DTOT + c * 32 + (hiH ? 8 : 0)];
            ((v4u*)&a)[0] = *(const v4u*)qp;
            ((v4u*)&a)[1] = *(const v4u*)(qp + 16);
            // B = K^T, 32x16 bf16 layout: lane<16 rows K=dims c*32+{0..7, 16..23}, lane>=16 rows {+8..15, +24..31}; col N=key=ln16
            const unsigned short* kp = &k_lds[ln16 * DTOT + c * 32 + (hiH ? 8 : 0)];
            ((v4u*)&bm)[0] = *(const v4u*)kp;
            ((v4u*)&bm)[1] = *(const v4u*)(kp + 16);
            s = __builtin_amdgcn_wmma_f32_16x16x32_bf16(false, a, false, bm, (short)0, s, false, false);
        }

        // ---- mask + online softmax (C layout: lane = key column, VGPR r = head r / r+8) ----
        const int  key_l = Idx[b * TOPK_N + kb * 16 + ln16];
        const bool valid = (key_l <= QSTART) && (key_l < SKV_N);

        #pragma unroll
        for (int r = 0; r < 8; ++r) {
            float sv   = valid ? s[r] : NEG_BIG;
            float mx   = redmax16(sv);
            float mnew = fmaxf(m_r[r], mx);
            float alpha = __expf(m_r[r] - mnew);
            float p     = __expf(sv - mnew);
            float rs    = redsum16(p);
            l_r[r] = l_r[r] * alpha + rs;
            m_r[r] = mnew;
            #pragma unroll
            for (int t = 0; t < 8; ++t) o_acc[t][r] *= alpha;
            const int m = r + (hiH ? 8 : 0);
            p_lds[wave * 256 + m * 16 + ln16] = f2bf(p);   // P[head][key], per-wave copy
        }

        // ---- PV: O[16h x 16d] += P(16x32, keys 16..31 zero) * V(32x16) ----
        v16bf pa;
        {
            const unsigned short* pp = &p_lds[wave * 256 + ln16 * 16];
            v4u z = {};
            ((v4u*)&pa)[0] = hiH ? z : *(const v4u*)pp;        // keys 0..15 on lanes 0..15
            ((v4u*)&pa)[1] = hiH ? z : *(const v4u*)(pp + 8);  // padded keys 16..31 (lanes 16..31) = 0
        }
        #pragma unroll
        for (int t = 0; t < 8; ++t) {
            v16bf vb;
            if (!hiH) {
                // rows K = keys 0..15 of column dim = wave*128 + t*16 + ln16
                const unsigned short* vp = &vt_lds[(wave * 128 + t * 16 + ln16) * 16];
                ((v4u*)&vb)[0] = *(const v4u*)vp;
                ((v4u*)&vb)[1] = *(const v4u*)(vp + 8);
            } else {
                v4u z = {};
                ((v4u*)&vb)[0] = z;                // rows K = keys 16..31 (don't exist) = 0
                ((v4u*)&vb)[1] = z;
            }
            o_acc[t] = __builtin_amdgcn_wmma_f32_16x16x32_bf16(false, pa, false, vb, (short)0, o_acc[t], false, false);
        }
        __syncthreads();   // protect k_lds / vt_lds before next gather
    }

    // ---- epilogue: O / l -> Out[b, 0, head, dim] ----
    float* outp = Out + (size_t)(b * HEADS + htile * 16) * DIM_V;
    #pragma unroll
    for (int r = 0; r < 8; ++r) {
        const int   m   = r + (hiH ? 8 : 0);
        const float inv = 1.0f / l_r[r];
        #pragma unroll
        for (int t = 0; t < 8; ++t) {
            outp[(size_t)m * DIM_V + wave * 128 + t * 16 + ln16] = o_acc[t][r] * inv;
        }
    }
}

extern "C" void kernel_launch(void* const* d_in, const int* in_sizes, int n_in,
                              void* d_out, int out_size, void* d_ws, size_t ws_size,
                              hipStream_t stream) {
    const float* Q   = (const float*)d_in[0];
    const float* KV  = (const float*)d_in[1];
    const int*   Idx = (const int*)d_in[2];
    float*       Out = (float*)d_out;

    dim3 grid(HEADS / 16, BATCH);   // (8, 32)
    dim3 block(128);
    sparse_mla_decode_kernel<<<grid, block, 0, stream>>>(Q, KV, Idx, Out);
}
